// SSClassifier_Cohan_Longformer_10651518894346
// MI455X (gfx1250) — compile-verified, matrix operand order checked by reading, not run on hardware
//
#include <hip/hip_runtime.h>
#include <hip/hip_bf16.h>

// ---------------- problem constants ----------------
#define B_    2
#define S_    4096
#define D_    768
#define H_    12
#define DH_   64
#define L_    2
#define HW_   256
#define C_    512
#define NC_   8
#define KW_   1024
#define NSEP_ 32
#define G_    33
#define NCLS_ 5
#define SEPID_ 2
#define NEG_  (-1000000000.0f)
#define EPS_  1e-5f

typedef __attribute__((ext_vector_type(16))) __bf16 v16bf;
typedef __attribute__((ext_vector_type(8)))  float  v8f;

// K-index inside a 16x32 bf16 A/B fragment (ISA 7.12.2):
// lanes 0-15 hold K={0..7,16..23}, lanes 16-31 hold K={8..15,24..31}
__device__ __forceinline__ int kidx(int j, int half) {
  return (j & 7) + ((j >> 3) << 4) + half * 8;
}

// ---------------- small utility kernels ----------------
__global__ void k_cvt_bf16(const float* __restrict__ src, __bf16* __restrict__ dst, long n) {
  long i = (long)blockIdx.x * blockDim.x + threadIdx.x;
  long st = (long)gridDim.x * blockDim.x;
  for (; i < n; i += st) dst[i] = (__bf16)src[i];
}

// transpose-convert: src is K x N (f32, row-major), dst is N x K (bf16, row-major)
__global__ void k_cvtT_bf16(const float* __restrict__ src, __bf16* __restrict__ dst,
                            int K, int N) {
  long n = (long)K * N;
  long i = (long)blockIdx.x * blockDim.x + threadIdx.x;
  long st = (long)gridDim.x * blockDim.x;
  for (; i < n; i += st) {
    long col = i / K, k = i - col * K;
    dst[i] = (__bf16)src[k * N + col];
  }
}

__global__ void k_gpos(const int* __restrict__ ids, int* __restrict__ gpos) {
  int b = blockIdx.x;
  if (threadIdx.x == 0) {
    gpos[b * G_] = 0;
    int cnt = 0;
    for (int s = 0; s < S_ && cnt < NSEP_; ++s)
      if (ids[b * S_ + s] == SEPID_) gpos[b * G_ + 1 + cnt++] = s;
    for (; cnt < NSEP_; ++cnt) gpos[b * G_ + 1 + cnt] = S_ - 1;
  }
}

// embedding + layernorm -> x (f32) and x (bf16)
__global__ __launch_bounds__(256) void k_embed_ln(
    const int* __restrict__ ids, const float* __restrict__ wemb,
    const float* __restrict__ pemb, const float* __restrict__ temb,
    const float* __restrict__ lnw, float* __restrict__ xo, __bf16* __restrict__ xbo) {
  int row = blockIdx.x;  // b*S + s
  int s = row % S_;
  int id = ids[row];
  __shared__ float buf[D_];
  __shared__ float red[256];
  float p1 = 0.f, p2 = 0.f;
  for (int d = threadIdx.x; d < D_; d += 256) {
    float v = wemb[(long)id * D_ + d] + pemb[(long)(s + 2) * D_ + d] + temb[d];
    buf[d] = v; p1 += v; p2 += v * v;
  }
  red[threadIdx.x] = p1; __syncthreads();
  for (int o = 128; o > 0; o >>= 1) { if (threadIdx.x < o) red[threadIdx.x] += red[threadIdx.x + o]; __syncthreads(); }
  float mean = red[0] / D_; __syncthreads();
  red[threadIdx.x] = p2; __syncthreads();
  for (int o = 128; o > 0; o >>= 1) { if (threadIdx.x < o) red[threadIdx.x] += red[threadIdx.x + o]; __syncthreads(); }
  float var = red[0] / D_ - mean * mean;
  float rstd = rsqrtf(var + EPS_);
  __syncthreads();
  for (int d = threadIdx.x; d < D_; d += 256) {
    float v = (buf[d] - mean) * rstd * lnw[d] + lnw[D_ + d];
    xo[(long)row * D_ + d] = v;
    xbo[(long)row * D_ + d] = (__bf16)v;
  }
}

// x = LN(x + a) -> f32 + bf16 (in-place safe: block owns its row)
__global__ __launch_bounds__(256) void k_add_ln(
    const float* __restrict__ x, const float* __restrict__ a,
    const float* __restrict__ lnw, float* __restrict__ xo, __bf16* __restrict__ xbo) {
  int row = blockIdx.x;
  __shared__ float buf[D_];
  __shared__ float red[256];
  float p1 = 0.f, p2 = 0.f;
  for (int d = threadIdx.x; d < D_; d += 256) {
    float v = x[(long)row * D_ + d] + a[(long)row * D_ + d];
    buf[d] = v; p1 += v; p2 += v * v;
  }
  red[threadIdx.x] = p1; __syncthreads();
  for (int o = 128; o > 0; o >>= 1) { if (threadIdx.x < o) red[threadIdx.x] += red[threadIdx.x + o]; __syncthreads(); }
  float mean = red[0] / D_; __syncthreads();
  red[threadIdx.x] = p2; __syncthreads();
  for (int o = 128; o > 0; o >>= 1) { if (threadIdx.x < o) red[threadIdx.x] += red[threadIdx.x + o]; __syncthreads(); }
  float var = red[0] / D_ - mean * mean;
  float rstd = rsqrtf(var + EPS_);
  __syncthreads();
  for (int d = threadIdx.x; d < D_; d += 256) {
    float v = (buf[d] - mean) * rstd * lnw[d] + lnw[D_ + d];
    xo[(long)row * D_ + d] = v;
    xbo[(long)row * D_ + d] = (__bf16)v;
  }
}

// ---------------- bf16 WMMA GEMM:  C(MxN) = A(MxK) * W^T(NxK) + bias ----------------
// Bw is the PRE-TRANSPOSED weight (N x K, row-major) so both A- and B-tiles are
// staged and fragment-loaded with contiguous ds_load_b128.
// block 256 thr = 8 waves (2x4), block tile 64x128, wave tile 32x32 (2x2 WMMA frags)
#define GBM 64
#define GBN 128
#define GBK 32
__global__ __launch_bounds__(256) void k_gemm_bf16(
    const __bf16* __restrict__ A, const __bf16* __restrict__ Bw,
    const float* __restrict__ bias, float* __restrict__ Cf,
    __bf16* __restrict__ Cb, int M, int N, int K, int act) {
  __shared__ __bf16 As[GBM][GBK + 8];   // 80B rows: 16B aligned, 20-dword stride
  __shared__ __bf16 Bs[GBN][GBK + 8];
  int m0 = blockIdx.x * GBM, n0 = blockIdx.y * GBN;
  int tid = threadIdx.x, lane = tid & 31, wid = tid >> 5;
  int wr = wid >> 2, wc = wid & 3;
  int half = lane >> 4, l16 = lane & 15;

  v8f zf = {};
  v8f c[2][2];
  c[0][0] = zf; c[0][1] = zf; c[1][0] = zf; c[1][1] = zf;

  int ar = tid >> 2, acc_ = (tid & 3) * 8;
  int bcol = tid >> 1, bko = (tid & 1) * 16;
  const __bf16* srcA0 = &A[(long)(m0 + ar) * K + acc_];
  const __bf16* srcB0 = &Bw[(long)(n0 + bcol) * K + bko];

  for (int k0 = 0; k0 < K; k0 += GBK) {
    __syncthreads();
    {
      const __bf16* srcA = srcA0 + k0;
#pragma unroll
      for (int j = 0; j < 8; ++j) As[ar][acc_ + j] = srcA[j];
      const __bf16* srcB = srcB0 + k0;
#pragma unroll
      for (int j = 0; j < 16; ++j) Bs[bcol][bko + j] = srcB[j];
      if (k0 + GBK < K) {                 // prefetch next K-tile (global_prefetch_b8)
        __builtin_prefetch(srcA + GBK, 0, 1);
        __builtin_prefetch(srcB + GBK, 0, 1);
      }
    }
    __syncthreads();

    v16bf af[2], bf2[2];
#pragma unroll
    for (int f = 0; f < 2; ++f) {
      int arow = wr * 32 + f * 16 + l16;
#pragma unroll
      for (int j = 0; j < 16; ++j) af[f][j] = As[arow][kidx(j, half)];
      int brow = wc * 32 + f * 16 + l16;
#pragma unroll
      for (int j = 0; j < 16; ++j) bf2[f][j] = Bs[brow][kidx(j, half)];
    }
#pragma unroll
    for (int fr = 0; fr < 2; ++fr)
#pragma unroll
      for (int fc = 0; fc < 2; ++fc)
        c[fr][fc] = __builtin_amdgcn_wmma_f32_16x16x32_bf16(
            false, af[fr], false, bf2[fc], (short)0, c[fr][fc], false, false);
  }

#pragma unroll
  for (int fr = 0; fr < 2; ++fr)
#pragma unroll
    for (int fc = 0; fc < 2; ++fc) {
      int col = n0 + wc * 32 + fc * 16 + l16;
      float bv = bias ? bias[col] : 0.f;
#pragma unroll
      for (int r = 0; r < 8; ++r) {
        int row = m0 + wr * 32 + fr * 16 + r + half * 8;
        float v = c[fr][fc][r] + bv;
        if (act == 1) v = 0.5f * v * (1.f + erff(v * 0.70710678118f));
        Cf[(long)row * N + col] = v;
        if (Cb) Cb[(long)row * N + col] = (__bf16)v;
      }
    }
}

// ---------------- flash-style windowed+global local attention ----------------
// block = (b, h, chunk nc, query-tile qt of 64 rows); 128 threads = 4 waves,
// each wave owns 16 query rows. Streams 1 padded global-key tile + 16 window
// key tiles of 64 keys; QK^T and P*V via v_wmma_f32_16x16x32_bf16.
// V is staged TRANSPOSED (vT[dh][key]) so P*V B-frags load contiguous b128.
__global__ __launch_bounds__(128) void k_local_attn(
    const float* __restrict__ q, const float* __restrict__ k,
    const float* __restrict__ v, const int* __restrict__ amask,
    const int* __restrict__ gpos, float* __restrict__ ctx) {
  __shared__ __bf16 q_s[64][72];    // 144B rows: 16B aligned, 36-dword stride
  __shared__ __bf16 k_s[64][72];
  __shared__ __bf16 vT_s[64][72];   // [dh][key]
  __shared__ __bf16 p_s[4][16][64]; // 128B rows
  __shared__ char kval_s[64];

  int bid = blockIdx.x;
  int qt = bid & 7;
  int nc = (bid >> 3) & 7;
  int h  = (bid >> 6) % H_;
  int b  = bid / (64 * H_);

  int tid = threadIdx.x, lane = tid & 31, w = tid >> 5;
  int half = lane >> 4, l16 = lane & 15;
  int qbase = nc * C_ + qt * 64;

  // stage Q (scaled by 1/sqrt(DH)) : each thread -> 32 dh of one row
  {
    int row = tid >> 1, dof = (tid & 1) * 32;
    const float* src = &q[((long)(b * S_ + qbase + row)) * D_ + h * DH_ + dof];
#pragma unroll
    for (int j = 0; j < 32; ++j) q_s[row][dof + j] = (__bf16)(src[j] * 0.125f);
  }
  __syncthreads();

  v16bf qa[2];
#pragma unroll
  for (int kk = 0; kk < 2; ++kk)
#pragma unroll
    for (int j = 0; j < 16; ++j) qa[kk][j] = q_s[w * 16 + l16][kk * 32 + kidx(j, half)];

  float m_run[8], l_run[8];
  v8f zf = {};
  v8f acc[4];
#pragma unroll
  for (int i = 0; i < 4; ++i) acc[i] = zf;
#pragma unroll
  for (int r = 0; r < 8; ++r) { m_run[r] = -1e30f; l_run[r] = 0.f; }

  for (int kt = -1; kt < 16; ++kt) {   // kt==-1 : padded global-token tile
    __syncthreads();
    {  // stage K (row-major) and V (transposed) tiles as bf16 + per-key validity
      int key = tid >> 1, dof = (tid & 1) * 32;
      long kp = 0; bool valid;
      if (kt < 0) {
        valid = key < G_;
        if (valid) kp = gpos[b * G_ + key];
      } else {
        int jw = kt * 64 + key;
        long p = (long)nc * C_ + jw - HW_;
        bool inb = (p >= 0 && p < S_);
        valid = inb;
        if (valid) valid = amask[b * S_ + p] > 0;
        kp = inb ? p : 0;
      }
      const float* ks = &k[((long)(b * S_) + kp) * D_ + h * DH_ + dof];
      const float* vs = &v[((long)(b * S_) + kp) * D_ + h * DH_ + dof];
#pragma unroll
      for (int j = 0; j < 32; ++j) {
        k_s[key][dof + j]  = valid ? (__bf16)ks[j] : (__bf16)0.f;
        vT_s[dof + j][key] = valid ? (__bf16)vs[j] : (__bf16)0.f;
      }
      if ((tid & 1) == 0) kval_s[key] = valid ? 1 : 0;
    }
    __syncthreads();

    // ---- scores: S = Q * K^T  (16 rows x 64 keys per wave) ----
    v8f sf[4];
#pragma unroll
    for (int i = 0; i < 4; ++i) sf[i] = zf;
#pragma unroll
    for (int nf = 0; nf < 4; ++nf)
#pragma unroll
      for (int kk = 0; kk < 2; ++kk) {
        v16bf bfrag;
#pragma unroll
        for (int j = 0; j < 16; ++j)
          bfrag[j] = k_s[nf * 16 + l16][kk * 32 + kidx(j, half)];
        sf[nf] = __builtin_amdgcn_wmma_f32_16x16x32_bf16(
            false, qa[kk], false, bfrag, (short)0, sf[nf], false, false);
      }

    // ---- masking + online softmax ----
    float sval[4][8];
#pragma unroll
    for (int nf = 0; nf < 4; ++nf) {
      int key = nf * 16 + l16;
      bool kv = kval_s[key] != 0;
      int jw = kt * 64 + key;
#pragma unroll
      for (int r = 0; r < 8; ++r) {
        int ci = qt * 64 + w * 16 + r + half * 8;  // row in chunk
        bool ok = kv;
        if (kt >= 0) {
          int rel = jw - ci;
          ok = ok && (rel >= 0) && (rel <= 2 * HW_);
        }
        sval[nf][r] = ok ? sf[nf][r] : NEG_;
      }
    }
    float mnew[8], tsum[8];
#pragma unroll
    for (int r = 0; r < 8; ++r) {
      float t = fmaxf(fmaxf(sval[0][r], sval[1][r]), fmaxf(sval[2][r], sval[3][r]));
      for (int m2 = 1; m2 < 16; m2 <<= 1) t = fmaxf(t, __shfl_xor(t, m2, 32));
      mnew[r] = fmaxf(m_run[r], t);
      tsum[r] = 0.f;
    }
#pragma unroll
    for (int nf = 0; nf < 4; ++nf)
#pragma unroll
      for (int r = 0; r < 8; ++r) {
        float p = __expf(sval[nf][r] - mnew[r]);
        tsum[r] += p;
        p_s[w][r + half * 8][nf * 16 + l16] = (__bf16)p;
      }
#pragma unroll
    for (int r = 0; r < 8; ++r) {
      float t = tsum[r];
      for (int m2 = 1; m2 < 16; m2 <<= 1) t += __shfl_xor(t, m2, 32);
      float alpha = __expf(m_run[r] - mnew[r]);
      l_run[r] = l_run[r] * alpha + t;
      m_run[r] = mnew[r];
#pragma unroll
      for (int nf = 0; nf < 4; ++nf) acc[nf][r] *= alpha;
    }
    __syncthreads();  // make p_s visible / fence staging reuse

    // ---- accumulate: acc += P * V ----
    v16bf pa[2];
#pragma unroll
    for (int kk = 0; kk < 2; ++kk)
#pragma unroll
      for (int j = 0; j < 16; ++j)
        pa[kk][j] = p_s[w][l16][kk * 32 + kidx(j, half)];
#pragma unroll
    for (int nf = 0; nf < 4; ++nf)
#pragma unroll
      for (int kk = 0; kk < 2; ++kk) {
        v16bf vb;
#pragma unroll
        for (int j = 0; j < 16; ++j)
          vb[j] = vT_s[nf * 16 + l16][kk * 32 + kidx(j, half)];
        acc[nf] = __builtin_amdgcn_wmma_f32_16x16x32_bf16(
            false, pa[kk], false, vb, (short)0, acc[nf], false, false);
      }
  }

  // write ctx in merged (B,S,D) layout
#pragma unroll
  for (int nf = 0; nf < 4; ++nf) {
    int dh = nf * 16 + l16;
#pragma unroll
    for (int r = 0; r < 8; ++r) {
      int sq = qbase + w * 16 + r + half * 8;
      ctx[((long)(b * S_ + sq)) * D_ + h * DH_ + dh] = acc[nf][r] / l_run[r];
    }
  }
}

// ---------------- global-token attention over full S (tiny FLOPs, VALU) ----------------
__global__ __launch_bounds__(256) void k_glob_attn(
    const float* __restrict__ qg, const float* __restrict__ kg,
    const float* __restrict__ vg, const int* __restrict__ amask,
    const int* __restrict__ gpos, float* __restrict__ og) {
  int bid = blockIdx.x;
  int g = bid % G_;
  int h = (bid / G_) % H_;
  int b = bid / (G_ * H_);
  __shared__ float sc[S_];
  __shared__ float qs[DH_];
  __shared__ float red[256];
  int tid = threadIdx.x;
  int pos = gpos[b * G_ + g];
  for (int d = tid; d < DH_; d += 256)
    qs[d] = qg[((long)(b * S_ + pos)) * D_ + h * DH_ + d] * 0.125f;
  __syncthreads();
  float lmax = -1e30f;
  for (int s = tid; s < S_; s += 256) {
    const float* kr = &kg[((long)(b * S_ + s)) * D_ + h * DH_];
    float dot = 0.f;
    for (int d = 0; d < DH_; ++d) dot += qs[d] * kr[d];
    if (!(amask[b * S_ + s] > 0)) dot = NEG_;
    sc[s] = dot;
    lmax = fmaxf(lmax, dot);
  }
  red[tid] = lmax; __syncthreads();
  for (int o = 128; o > 0; o >>= 1) { if (tid < o) red[tid] = fmaxf(red[tid], red[tid + o]); __syncthreads(); }
  float M = red[0]; __syncthreads();
  float lsum = 0.f;
  for (int s = tid; s < S_; s += 256) { float p = __expf(sc[s] - M); sc[s] = p; lsum += p; }
  red[tid] = lsum; __syncthreads();
  for (int o = 128; o > 0; o >>= 1) { if (tid < o) red[tid] += red[tid + o]; __syncthreads(); }
  float denom = red[0]; __syncthreads();
  int d = tid & 63, grp = tid >> 6;
  float part = 0.f;
  for (int s = grp; s < S_; s += 4)
    part += sc[s] * vg[((long)(b * S_ + s)) * D_ + h * DH_ + d];
  red[tid] = part; __syncthreads();
  if (grp == 0) {
    float t = red[tid] + red[tid + 64] + red[tid + 128] + red[tid + 192];
    og[(((long)b * H_ + h) * G_ + g) * DH_ + d] = t / denom;
  }
}

// overwrite global rows of ctx with og
__global__ void k_merge_global(const float* __restrict__ og,
                               const int* __restrict__ gpos, float* __restrict__ ctx) {
  int bg = blockIdx.x;
  int b = bg / G_, g = bg % G_;
  int row = gpos[b * G_ + g];
  for (int d = threadIdx.x; d < D_; d += 256) {
    int h = d >> 6, dh = d & 63;
    ctx[((long)(b * S_ + row)) * D_ + d] = og[(((long)b * H_ + h) * G_ + g) * DH_ + dh];
  }
}

// gather SEP rows -> classifier head (64 x 768) @ (768 x 5)
__global__ __launch_bounds__(256) void k_cls(
    const float* __restrict__ x, const int* __restrict__ gpos,
    const float* __restrict__ W, const float* __restrict__ bias, float* __restrict__ out) {
  int rid = blockIdx.x;  // b*NSEP + j
  int b = rid / NSEP_, j = rid % NSEP_;
  int pos = gpos[b * G_ + 1 + j];
  const float* xr = &x[((long)(b * S_ + pos)) * D_];
  __shared__ float red[256];
  for (int c = 0; c < NCLS_; ++c) {
    float part = 0.f;
    for (int d = threadIdx.x; d < D_; d += 256) part += xr[d] * W[(long)c * D_ + d];
    red[threadIdx.x] = part; __syncthreads();
    for (int o = 128; o > 0; o >>= 1) { if (threadIdx.x < o) red[threadIdx.x] += red[threadIdx.x + o]; __syncthreads(); }
    if (threadIdx.x == 0) out[(long)rid * NCLS_ + c] = red[0] + bias[c];
    __syncthreads();
  }
}

// ---------------- host orchestration ----------------
extern "C" void kernel_launch(void* const* d_in, const int* in_sizes, int n_in,
                              void* d_out, int out_size, void* d_ws, size_t ws_size,
                              hipStream_t stream) {
  const int*   ids   = (const int*)d_in[0];
  const int*   amask = (const int*)d_in[1];
  const float* wemb  = (const float*)d_in[2];
  const float* pemb  = (const float*)d_in[3];
  const float* temb  = (const float*)d_in[4];
  const float* embln = (const float*)d_in[5];
  const float* Wqkv  = (const float*)d_in[6];
  const float* bqkv  = (const float*)d_in[7];
  const float* Wqkvg = (const float*)d_in[8];
  const float* bqkvg = (const float*)d_in[9];
  const float* Wo    = (const float*)d_in[10];
  const float* bo    = (const float*)d_in[11];
  const float* ln1   = (const float*)d_in[12];
  const float* Wff1  = (const float*)d_in[13];
  const float* bff1  = (const float*)d_in[14];
  const float* Wff2  = (const float*)d_in[15];
  const float* bff2  = (const float*)d_in[16];
  const float* ln2   = (const float*)d_in[17];
  const float* clsW  = (const float*)d_in[18];
  const float* clsb  = (const float*)d_in[19];

  char* ws = (char*)d_ws;
  size_t off = 0;
  auto alloc = [&](size_t bytes) -> char* {
    char* p = ws + off;
    off = (off + bytes + 255) & ~(size_t)255;
    return p;
  };
  const long BSD = (long)B_ * S_ * D_;
  int*    gpos    = (int*)alloc((size_t)B_ * G_ * sizeof(int));
  float*  x       = (float*)alloc(BSD * 4);
  __bf16* xbf     = (__bf16*)alloc(BSD * 2);
  float*  qb      = (float*)alloc(BSD * 4);
  float*  kb      = (float*)alloc(BSD * 4);
  float*  vb      = (float*)alloc(BSD * 4);
  float*  qg      = (float*)alloc(BSD * 4);
  float*  kg      = (float*)alloc(BSD * 4);
  float*  vg      = (float*)alloc(BSD * 4);
  float*  ctx     = (float*)alloc(BSD * 4);
  __bf16* ctxbf   = (__bf16*)alloc(BSD * 2);
  float*  tmp     = (float*)alloc(BSD * 4);          // attn-proj / ff2 output
  float*  h1      = (float*)alloc(BSD * 4 * 4);      // B*S*4D
  __bf16* h1bf    = (__bf16*)alloc(BSD * 4 * 2);
  float*  og      = (float*)alloc((long)B_ * H_ * G_ * DH_ * 4);
  const long wqkv_n = (long)L_ * 3 * D_ * D_;
  __bf16* Wqkv_bf  = (__bf16*)alloc(wqkv_n * 2);      // each D x D block transposed
  __bf16* Wqkvg_bf = (__bf16*)alloc(wqkv_n * 2);
  __bf16* Wo_bf    = (__bf16*)alloc((long)L_ * D_ * D_ * 2);
  __bf16* Wff1_bf  = (__bf16*)alloc((long)L_ * D_ * 4 * D_ * 2);
  __bf16* Wff2_bf  = (__bf16*)alloc((long)L_ * 4 * D_ * D_ * 2);

  auto cvt = [&](const float* s, __bf16* d, long n) {
    long nb = (n + 255) / 256; if (nb > 4096) nb = 4096;
    k_cvt_bf16<<<dim3((unsigned)nb), dim3(256), 0, stream>>>(s, d, n);
  };
  // transpose-convert one K x N weight matrix into N x K bf16
  auto cvtT = [&](const float* s, __bf16* d, int K, int N) {
    long n = (long)K * N;
    long nb = (n + 255) / 256; if (nb > 4096) nb = 4096;
    k_cvtT_bf16<<<dim3((unsigned)nb), dim3(256), 0, stream>>>(s, d, K, N);
  };
  for (int i = 0; i < L_ * 3; ++i) {
    cvtT(Wqkv  + (long)i * D_ * D_, Wqkv_bf  + (long)i * D_ * D_, D_, D_);
    cvtT(Wqkvg + (long)i * D_ * D_, Wqkvg_bf + (long)i * D_ * D_, D_, D_);
  }
  for (int l = 0; l < L_; ++l) {
    cvtT(Wo   + (long)l * D_ * D_,     Wo_bf   + (long)l * D_ * D_,     D_, D_);
    cvtT(Wff1 + (long)l * D_ * 4 * D_, Wff1_bf + (long)l * D_ * 4 * D_, D_, 4 * D_);
    cvtT(Wff2 + (long)l * 4 * D_ * D_, Wff2_bf + (long)l * 4 * D_ * D_, 4 * D_, D_);
  }

  k_gpos<<<dim3(B_), dim3(32), 0, stream>>>(ids, gpos);
  k_embed_ln<<<dim3(B_ * S_), dim3(256), 0, stream>>>(ids, wemb, pemb, temb, embln, x, xbf);

  auto gemm = [&](const __bf16* A, const __bf16* Wt, const float* bias,
                  float* Cf, __bf16* Cb, int M, int N, int K, int act) {
    dim3 grid(M / GBM, N / GBN);
    k_gemm_bf16<<<grid, dim3(256), 0, stream>>>(A, Wt, bias, Cf, Cb, M, N, K, act);
  };

  const int M = B_ * S_;
  for (int l = 0; l < L_; ++l) {
    const __bf16* wq = Wqkv_bf  + (long)l * 3 * D_ * D_;
    const __bf16* wg = Wqkvg_bf + (long)l * 3 * D_ * D_;
    gemm(xbf, wq,                     bqkv  + (long)(l * 3 + 0) * D_, qb, nullptr, M, D_, D_, 0);
    gemm(xbf, wq + (long)1 * D_ * D_, bqkv  + (long)(l * 3 + 1) * D_, kb, nullptr, M, D_, D_, 0);
    gemm(xbf, wq + (long)2 * D_ * D_, bqkv  + (long)(l * 3 + 2) * D_, vb, nullptr, M, D_, D_, 0);
    gemm(xbf, wg,                     bqkvg + (long)(l * 3 + 0) * D_, qg, nullptr, M, D_, D_, 0);
    gemm(xbf, wg + (long)1 * D_ * D_, bqkvg + (long)(l * 3 + 1) * D_, kg, nullptr, M, D_, D_, 0);
    gemm(xbf, wg + (long)2 * D_ * D_, bqkvg + (long)(l * 3 + 2) * D_, vg, nullptr, M, D_, D_, 0);

    k_local_attn<<<dim3(B_ * H_ * NC_ * 8), dim3(128), 0, stream>>>(qb, kb, vb, amask, gpos, ctx);
    k_glob_attn<<<dim3(B_ * H_ * G_), dim3(256), 0, stream>>>(qg, kg, vg, amask, gpos, og);
    k_merge_global<<<dim3(B_ * G_), dim3(256), 0, stream>>>(og, gpos, ctx);
    cvt(ctx, ctxbf, BSD);

    gemm(ctxbf, Wo_bf + (long)l * D_ * D_, bo + (long)l * D_, tmp, nullptr, M, D_, D_, 0);
    k_add_ln<<<dim3(M), dim3(256), 0, stream>>>(x, tmp, ln1 + (long)l * 2 * D_, x, xbf);

    gemm(xbf, Wff1_bf + (long)l * D_ * 4 * D_, bff1 + (long)l * 4 * D_, h1, h1bf, M, 4 * D_, D_, 1);
    gemm(h1bf, Wff2_bf + (long)l * 4 * D_ * D_, bff2 + (long)l * D_, tmp, nullptr, M, D_, 4 * D_, 0);
    k_add_ln<<<dim3(M), dim3(256), 0, stream>>>(x, tmp, ln2 + (long)l * 2 * D_, x, xbf);
  }

  k_cls<<<dim3(B_ * NSEP_), dim3(256), 0, stream>>>(x, gpos, clsW, clsb, (float*)d_out);
}
